// FeedForward_62294205661881
// MI455X (gfx1250) — 8-bit matrix operand order repaired and checked by reading, compiled, not run on hardware
//
#include <hip/hip_runtime.h>
#include <hip/hip_bf16.h>
#include <stdint.h>


typedef __attribute__((ext_vector_type(2))) int v2i;
typedef __attribute__((ext_vector_type(4))) int v4i;
typedef __attribute__((ext_vector_type(8))) int v8i;

#define EPSV 1e-5f

#if __has_builtin(__builtin_amdgcn_global_load_async_to_lds_b128)
#define ASYNC_LDS 1
typedef __attribute__((address_space(1))) v4i gv4i;
typedef __attribute__((address_space(3))) v4i lv4i;
__device__ __forceinline__ void async_copy16(void* lds, const void* g) {
  __builtin_amdgcn_global_load_async_to_lds_b128((gv4i*)g, (lv4i*)lds, 0, 0);
}
#if __has_builtin(__builtin_amdgcn_s_wait_asynccnt)
#define WAIT_ASYNC(n) __builtin_amdgcn_s_wait_asynccnt(n)
#else
#define WAIT_ASYNC(n) asm volatile("s_wait_asynccnt %0" ::"i"(n) : "memory")
#endif
#else
#define ASYNC_LDS 0
#endif

__global__ __launch_bounds__(256) void absmean_partial(
    const float* __restrict__ W, size_t n, float* __restrict__ partial) {
  __shared__ float red[256];
  float s = 0.0f;
  for (size_t i = (size_t)blockIdx.x * 256 + threadIdx.x; i < n;
       i += (size_t)gridDim.x * 256)
    s += fabsf(W[i]);
  red[threadIdx.x] = s;
  __syncthreads();
  for (int o = 128; o > 0; o >>= 1) {
    if ((int)threadIdx.x < o) red[threadIdx.x] += red[threadIdx.x + o];
    __syncthreads();
  }
  if (threadIdx.x == 0) partial[blockIdx.x] = red[0];
}

__global__ __launch_bounds__(256) void absmean_final(
    const float* __restrict__ partial, int np, float invn,
    float* __restrict__ gamma) {
  __shared__ float red[256];
  float s = 0.0f;
  for (int i = threadIdx.x; i < np; i += 256) s += partial[i];
  red[threadIdx.x] = s;
  __syncthreads();
  for (int o = 128; o > 0; o >>= 1) {
    if ((int)threadIdx.x < o) red[threadIdx.x] += red[threadIdx.x + o];
    __syncthreads();
  }
  if (threadIdx.x == 0) *gamma = red[0] * invn + EPSV;
}

__global__ __launch_bounds__(256) void quant_weight_k(
    const float* __restrict__ W, const float* __restrict__ gamma, size_t n,
    int8_t* __restrict__ Wq) {
  const float inv = 1.0f / (*gamma);
  for (size_t i = (size_t)blockIdx.x * blockDim.x + threadIdx.x; i < n;
       i += (size_t)gridDim.x * blockDim.x) {
    float q = rintf(W[i] * inv);
    q = fminf(fmaxf(q, -1.0f), 1.0f);
    Wq[i] = (int8_t)q;
  }
}

__global__ __launch_bounds__(256) void quant_rows_k(
    const float* __restrict__ X, int K, int8_t* __restrict__ Xq,
    float* __restrict__ scale) {
  __shared__ float red[256];
  const size_t row = blockIdx.x;
  const float* Xr = X + row * (size_t)K;
  float m = 0.0f;
  for (int i = threadIdx.x * 4; i < K; i += 1024) {
    const float4 v = *(const float4*)&Xr[i];
    m = fmaxf(m, fmaxf(fmaxf(fabsf(v.x), fabsf(v.y)),
                       fmaxf(fabsf(v.z), fabsf(v.w))));
  }
  red[threadIdx.x] = m;
  __syncthreads();
  for (int o = 128; o > 0; o >>= 1) {
    if ((int)threadIdx.x < o)
      red[threadIdx.x] = fmaxf(red[threadIdx.x], red[threadIdx.x + o]);
    __syncthreads();
  }
  const float s = 127.0f / fmaxf(red[0], EPSV);
  if (threadIdx.x == 0) scale[row] = s;
  int8_t* Xqr = Xq + row * (size_t)K;
  for (int i = threadIdx.x * 4; i < K; i += 1024) {
    const float4 v = *(const float4*)&Xr[i];
    char4 q;
    q.x = (int8_t)fminf(fmaxf(rintf(v.x * s), -128.0f), 127.0f);
    q.y = (int8_t)fminf(fmaxf(rintf(v.y * s), -128.0f), 127.0f);
    q.z = (int8_t)fminf(fmaxf(rintf(v.z * s), -128.0f), 127.0f);
    q.w = (int8_t)fminf(fmaxf(rintf(v.w * s), -128.0f), 127.0f);
    *(char4*)&Xqr[i] = q;
  }
}

__global__ __launch_bounds__(256) void bitnet_gemm(
    const int8_t* __restrict__ Aq, const float* __restrict__ aScale,
    const int8_t* __restrict__ Wq, const float* __restrict__ gamma,
    float* __restrict__ Out, int M, int N, int K, int relu2) {
  __shared__ __align__(16) int8_t lA[2][128][64];
  __shared__ __align__(16) int8_t lB[2][128][64];

  const int tid = threadIdx.x;
  const int wv = tid >> 5;
  const int l = tid & 31;
  const int li = l & 15;
  const int lh = l >> 4;
  const int wvM = wv & 3;
  const int wvN = wv >> 2;
  const int mBlk = blockIdx.y * 128;
  const int nBlk = blockIdx.x * 128;

  const int r0 = tid >> 2, c0 = (tid & 3) << 4;
  const int r1 = (tid + 256) >> 2, c1 = c0;
  const int8_t* gA0 = &Aq[(size_t)(mBlk + r0) * K + c0];
  const int8_t* gA1 = &Aq[(size_t)(mBlk + r1) * K + c1];
  const int8_t* gB0 = &Wq[(size_t)(nBlk + r0) * K + c0];
  const int8_t* gB1 = &Wq[(size_t)(nBlk + r1) * K + c1];

  v8i acc[2][4];
#pragma unroll
  for (int mi = 0; mi < 2; ++mi)
#pragma unroll
    for (int t = 0; t < 4; ++t) acc[mi][t] = (v8i)0;

  const int nSteps = K >> 6;
  const int lastK = (nSteps - 1) << 6;

#if ASYNC_LDS
  async_copy16(&lA[0][r0][c0], gA0 + 0);
  async_copy16(&lA[0][r1][c1], gA1 + 0);
  async_copy16(&lB[0][r0][c0], gB0 + 0);
  async_copy16(&lB[0][r1][c1], gB1 + 0);
#else
  int4 sA0, sA1, sB0, sB1;
  sA0 = *(const int4*)(gA0);
  sA1 = *(const int4*)(gA1);
  sB0 = *(const int4*)(gB0);
  sB1 = *(const int4*)(gB1);
  *(int4*)&lA[0][r0][c0] = sA0;
  *(int4*)&lA[0][r1][c1] = sA1;
  *(int4*)&lB[0][r0][c0] = sB0;
  *(int4*)&lB[0][r1][c1] = sB1;
#endif

  for (int s = 0; s < nSteps; ++s) {
    const int cur = s & 1;
    const int nk = (s + 1 < nSteps) ? ((s + 1) << 6) : lastK;

#if ASYNC_LDS
    async_copy16(&lA[cur ^ 1][r0][c0], gA0 + nk);
    async_copy16(&lA[cur ^ 1][r1][c1], gA1 + nk);
    async_copy16(&lB[cur ^ 1][r0][c0], gB0 + nk);
    async_copy16(&lB[cur ^ 1][r1][c1], gB1 + nk);
    WAIT_ASYNC(4);
#else
    sA0 = *(const int4*)(gA0 + nk);
    sA1 = *(const int4*)(gA1 + nk);
    sB0 = *(const int4*)(gB0 + nk);
    sB1 = *(const int4*)(gB1 + nk);
#endif
    __syncthreads();

    v8i a[2];
#pragma unroll
    for (int mi = 0; mi < 2; ++mi) {
      const int8_t* ap = &lA[cur][wvM * 32 + mi * 16 + li][lh * 16];
      const v2i a0 = *(const v2i*)(ap);
      const v2i a1 = *(const v2i*)(ap + 8);
      const v2i a2 = *(const v2i*)(ap + 32);
      const v2i a3 = *(const v2i*)(ap + 40);
      a[mi] = (v8i){a0.x, a0.y, a1.x, a1.y, a2.x, a2.y, a3.x, a3.y};
    }
#pragma unroll
    for (int tN = 0; tN < 4; ++tN) {
      const int8_t* bp = &lB[cur][wvN * 64 + tN * 16 + li][lh * 16];
      const v4i b0 = *(const v4i*)(bp);
      const v4i b1 = *(const v4i*)(bp + 32);
      const v8i b = (v8i){b0.x, b0.y, b0.z, b0.w, b1.x, b1.y, b1.z, b1.w};
      acc[0][tN] = __builtin_amdgcn_wmma_i32_16x16x64_iu8(
          true, a[0], true, b, acc[0][tN], false, false);
      acc[1][tN] = __builtin_amdgcn_wmma_i32_16x16x64_iu8(
          true, a[1], true, b, acc[1][tN], false, false);
    }
    __syncthreads();

#if !ASYNC_LDS
    *(int4*)&lA[cur ^ 1][r0][c0] = sA0;
    *(int4*)&lA[cur ^ 1][r1][c1] = sA1;
    *(int4*)&lB[cur ^ 1][r0][c0] = sB0;
    *(int4*)&lB[cur ^ 1][r1][c1] = sB1;
#endif
  }

  const float ga = *gamma;
#pragma unroll
  for (int mi = 0; mi < 2; ++mi) {
    const int mBase = mBlk + wvM * 32 + mi * 16 + lh * 8;
#pragma unroll
    for (int r = 0; r < 8; ++r) {
      const int m = mBase + r;
      const float f = ga / aScale[m];
#pragma unroll
      for (int tN = 0; tN < 4; ++tN) {
        const int n = nBlk + wvN * 64 + tN * 16 + li;
        float v = (float)acc[mi][tN][r] * f;
        if (relu2) { v = fmaxf(v, 0.0f); v = v * v; }
        Out[(size_t)m * N + n] = v;
      }
    }
  }
}

extern "C" void kernel_launch(void* const* d_in, const int* in_sizes, int n_in,
                              void* d_out, int out_size, void* d_ws,
                              size_t ws_size, hipStream_t stream) {
  const float* x  = (const float*)d_in[0];
  const float* w1 = (const float*)d_in[1];
  const float* w2 = (const float*)d_in[2];
  float* out = (float*)d_out;

  const int d_model = 2048;
  const int d_ff    = 8192;
  const int M       = in_sizes[0] / d_model;

  uint8_t* ws = (uint8_t*)d_ws;
  int8_t* xq  = (int8_t*)ws;
  int8_t* w1q = xq + (size_t)M * d_model;
  int8_t* w2q = w1q + (size_t)d_ff * d_model;
  int8_t* hq  = w2q + (size_t)d_model * d_ff;
  float*  h   = (float*)(hq + (size_t)M * d_ff);
  float* xscale  = h + (size_t)M * d_ff;
  float* hscale  = xscale + M;
  float* g1      = hscale + M;
  float* g2      = g1 + 1;
  float* partial = g2 + 1;

  const size_t nW = (size_t)d_ff * d_model;

  absmean_partial<<<512, 256, 0, stream>>>(w1, nW, partial);
  absmean_final<<<1, 256, 0, stream>>>(partial, 512, 1.0f / (float)nW, g1);
  quant_weight_k<<<2048, 256, 0, stream>>>(w1, g1, nW, w1q);

  absmean_partial<<<512, 256, 0, stream>>>(w2, nW, partial);
  absmean_final<<<1, 256, 0, stream>>>(partial, 512, 1.0f / (float)nW, g2);
  quant_weight_k<<<2048, 256, 0, stream>>>(w2, g2, nW, w2q);

  quant_rows_k<<<M, 256, 0, stream>>>(x, d_model, xq, xscale);

  dim3 g1g(d_ff / 128, M / 128);
  bitnet_gemm<<<g1g, 256, 0, stream>>>(xq, xscale, w1q, g1, h,
                                       M, d_ff, d_model,  1);

  quant_rows_k<<<M, 256, 0, stream>>>(h, d_ff, hq, hscale);

  dim3 g2g(d_model / 128, M / 128);
  bitnet_gemm<<<g2g, 256, 0, stream>>>(hq, hscale, w2q, g2, out,
                                       M, d_model, d_ff,  0);
}
